// HybridRC_35442070126738
// MI455X (gfx1250) — matrix operand order repaired and checked by reading, not compiled, not run on hardware
//
#include <hip/hip_runtime.h>
#include <cmath>

// ---------------------------------------------------------------------------
// Problem constants (from reference)
// ---------------------------------------------------------------------------
#define BATCH   64
#define FRAMES  128
#define N_IN    512
#define N_HID   2048
#define MROWS   (BATCH * FRAMES)          // 8192 rows of x (b*128 + t)
#define ALPHA   0.5f
#define DECAY   0.5f
#define THR     0.5f

// d_out layout: mems [2][64][129][2048] then spikes [2][64][129][2048]
#define SPK_OFF ((size_t)2 * BATCH * (FRAMES + 1) * N_HID)   // 33,816,576

// Workspace byte offsets (all 256B-aligned)
#define XB_OFF   ((size_t)0)                                 // x bf16 [8192][512]
#define W1T_OFF  (XB_OFF  + (size_t)MROWS * N_IN * 2)        // W_in1^T bf16 [2048][512]
#define W2T_OFF  (W1T_OFF + (size_t)N_HID * N_IN * 2)
#define A1T_OFF  (W2T_OFF + (size_t)N_HID * N_IN * 2)        // A1^T bf16 [2048][2048]
#define A2T_OFF  (A1T_OFF + (size_t)N_HID * N_HID * 2)
#define U_OFF    (A2T_OFF + (size_t)N_HID * N_HID * 2)       // U f32 [2][8192][2048]
#define S0_OFF   (U_OFF   + (size_t)2 * MROWS * N_HID * 4)   // spike ping bf16 [2][64][2048]
#define S1_OFF   (S0_OFF  + (size_t)2 * BATCH * N_HID * 2)   // spike pong

// ---------------------------------------------------------------------------
// Types for WMMA bf16
// ---------------------------------------------------------------------------
typedef __attribute__((ext_vector_type(16))) __bf16         v16bf;
typedef __attribute__((ext_vector_type(8)))  float          v8f;
typedef __attribute__((ext_vector_type(8)))  unsigned short v8u;
typedef __attribute__((ext_vector_type(16))) unsigned short v16u;

__device__ __forceinline__ unsigned short f2bf(float f) {
    unsigned int u = __builtin_bit_cast(unsigned int, f);
    u += 0x7FFFu + ((u >> 16) & 1u);                 // round-to-nearest-even
    return (unsigned short)(u >> 16);
}

// Build one 16-element bf16 fragment from two contiguous 16-byte chunks.
__device__ __forceinline__ v16bf load_frag(const unsigned short* p0,
                                           const unsigned short* p1) {
    v8u lo = *(const v8u*)p0;
    v8u hi = *(const v8u*)p1;
    v16u c = __builtin_shufflevector(lo, hi,
                                     0, 1, 2, 3, 4, 5, 6, 7,
                                     8, 9, 10, 11, 12, 13, 14, 15);
    return __builtin_bit_cast(v16bf, c);
}

__device__ __forceinline__ v8f wmma_bf16(v16bf a, v16bf b, v8f c) {
    return __builtin_amdgcn_wmma_f32_16x16x32_bf16(
        false, a, false, b, (short)0, c, false, false);
}

// ---------------------------------------------------------------------------
// Setup: f32 -> bf16 convert (row-major copy) and transposing convert
// ---------------------------------------------------------------------------
__global__ void k_cvt(const float* __restrict__ in, unsigned short* __restrict__ out, int n) {
    int i = blockIdx.x * blockDim.x + threadIdx.x;
    if (i < n) out[i] = f2bf(in[i]);
}

// in: f32 [R][C] row-major ; out: bf16 [C][R]  (i.e. out holds in^T)
__global__ void k_transpose(const float* __restrict__ in, unsigned short* __restrict__ out,
                            int R, int C) {
    int i = blockIdx.x * blockDim.x + threadIdx.x;
    if (i < R * C) {
        int r = i / C, c = i % C;
        out[(size_t)c * R + r] = f2bf(in[i]);
    }
}

// ---------------------------------------------------------------------------
// U = x @ W_in + b   (M=8192, K=512, N=2048, per layer)
// Block: 256 thr = 8 waves, arranged 2 M-tiles x 4 N-groups; wave tile 16x64.
// Grid: (2048/256, 8192/32, 2 layers). Compiler-pipelined K loop.
// ---------------------------------------------------------------------------
__global__ void __launch_bounds__(256) k_gemm_u(
    const unsigned short* __restrict__ xb,    // bf16 [8192][512]
    const unsigned short* __restrict__ w1t,   // bf16 [2048][512]
    const unsigned short* __restrict__ w2t,
    const float* __restrict__ bias1,
    const float* __restrict__ bias2,
    float* __restrict__ U)                    // f32 [2][8192][2048]
{
    const int lane = threadIdx.x & 31;
    const int w    = threadIdx.x >> 5;
    const int l    = blockIdx.z;
    const unsigned short* wt   = l ? w2t : w1t;
    const float*          bias = l ? bias2 : bias1;

    const int m0   = blockIdx.y * 32 + (w >> 2) * 16;
    const int n0   = blockIdx.x * 256 + (w & 3) * 64;
    const int half = lane >> 4;     // 0 | 1
    const int lrow = lane & 15;

    const unsigned short* arow = xb + (size_t)(m0 + lrow) * N_IN + half * 8;
    const unsigned short* brow[4];
#pragma unroll
    for (int j = 0; j < 4; ++j)
        brow[j] = wt + (size_t)(n0 + j * 16 + lrow) * N_IN + half * 8;

    v8f acc[4] = {v8f{}, v8f{}, v8f{}, v8f{}};

#pragma unroll 2
    for (int kk = 0; kk < N_IN; kk += 32) {
        v16bf a = load_frag(arow + kk, arow + kk + 16);
#pragma unroll
        for (int j = 0; j < 4; ++j) {
            v16bf b = load_frag(brow[j] + kk, brow[j] + kk + 16);
            acc[j] = wmma_bf16(a, b, acc[j]);
        }
    }

    float* Ul = U + (size_t)l * MROWS * N_HID;
#pragma unroll
    for (int j = 0; j < 4; ++j) {
        const int n = n0 + j * 16 + lrow;
        const float bv = bias[n];
#pragma unroll
        for (int r = 0; r < 8; ++r) {
            const int m = m0 + r + half * 8;
            Ul[(size_t)m * N_HID + n] = acc[j][r] + bv;
        }
    }
}

// ---------------------------------------------------------------------------
// Init: t=0 membranes/spikes in d_out, zero bf16 spike ping buffer
// ---------------------------------------------------------------------------
__global__ void k_init(const float* __restrict__ m01, const float* __restrict__ m02,
                       float* __restrict__ out, unsigned short* __restrict__ s0)
{
    int i = blockIdx.x * blockDim.x + threadIdx.x;      // over 2*64*2048
    if (i >= 2 * BATCH * N_HID) return;
    int l   = i / (BATCH * N_HID);
    int rem = i % (BATCH * N_HID);
    int b   = rem / N_HID;
    int h   = rem % N_HID;
    const float* m0p = l ? m02 : m01;
    size_t idx = (((size_t)l * BATCH + b) * (FRAMES + 1)) * N_HID + h;   // t = 0
    out[idx]           = m0p[rem];
    out[SPK_OFF + idx] = 0.0f;
    s0[i] = 0;      // s0 bf16 == 0.0
}

// ---------------------------------------------------------------------------
// One recurrent step: y = tanh(0.5*(s@A) + 0.5*u), LIF update, spike.
// GEMM M=64, K=2048, N=2048 per layer via bf16 WMMA.
// Block: 8 waves = 4 M-tiles x 2 N-groups; wave tile 16x32 (2 accumulators)
// -> 512 waves/step for latency hiding in this tiny, latency-bound GEMM.
// Grid: (2048/64, 1, 2 layers). Compiler-pipelined K loop.
// ---------------------------------------------------------------------------
__global__ void __launch_bounds__(256) k_step(
    const unsigned short* __restrict__ a1t,   // bf16 A1^T [2048][2048]
    const unsigned short* __restrict__ a2t,
    const float* __restrict__ U,              // f32 [2][8192][2048]
    const unsigned short* __restrict__ s_in,  // bf16 [2][64][2048]
    unsigned short* __restrict__ s_out,       // bf16 [2][64][2048]
    float* __restrict__ out,                  // d_out
    int t)
{
    const int lane = threadIdx.x & 31;
    const int w    = threadIdx.x >> 5;
    const int l    = blockIdx.z;
    const unsigned short* at = l ? a2t : a1t;

    const int m0   = (w >> 1) * 16;                      // 0..48 (batch rows)
    const int n0   = blockIdx.x * 64 + (w & 1) * 32;     // hidden cols
    const int half = lane >> 4;
    const int lrow = lane & 15;

    const unsigned short* srow =
        s_in + (size_t)l * BATCH * N_HID + (size_t)(m0 + lrow) * N_HID + half * 8;
    const unsigned short* brow[2];
#pragma unroll
    for (int j = 0; j < 2; ++j)
        brow[j] = at + (size_t)(n0 + j * 16 + lrow) * N_HID + half * 8;

    v8f acc[2] = {v8f{}, v8f{}};

#pragma unroll 2
    for (int kk = 0; kk < N_HID; kk += 32) {
        v16bf a = load_frag(srow + kk, srow + kk + 16);
#pragma unroll
        for (int j = 0; j < 2; ++j) {
            v16bf b = load_frag(brow[j] + kk, brow[j] + kk + 16);
            acc[j] = wmma_bf16(a, b, acc[j]);
        }
    }

    // Fused LIF epilogue straight out of the accumulators.
#pragma unroll
    for (int j = 0; j < 2; ++j) {
        const int h = n0 + j * 16 + lrow;
#pragma unroll
        for (int r = 0; r < 8; ++r) {
            const int b = m0 + r + half * 8;              // batch index 0..63
            const float dot = acc[j][r];
            const float u   =
                U[((size_t)l * MROWS + (size_t)b * FRAMES + t) * N_HID + h];
            const float y = tanhf(ALPHA * dot + (1.0f - ALPHA) * u);

            const size_t sidx  = ((size_t)l * BATCH + b) * N_HID + h;
            const size_t row0  = ((size_t)l * BATCH + b) * (FRAMES + 1);
            const size_t mprev = (row0 + t) * N_HID + h;
            const size_t mnext = (row0 + t + 1) * N_HID + h;

            const float s_old = s_in[sidx] ? 1.0f : 0.0f;
            const float m_old = out[mprev];
            const float m_new = m_old * DECAY - THR * (1.0f - s_old) + y;
            const float s_new = (m_new > THR) ? 1.0f : 0.0f;

            out[mnext]           = m_new;
            out[SPK_OFF + mnext] = s_new;
            s_out[sidx] = (m_new > THR) ? (unsigned short)0x3F80u
                                        : (unsigned short)0u;
        }
    }
}

// ---------------------------------------------------------------------------
// Host launcher
// ---------------------------------------------------------------------------
extern "C" void kernel_launch(void* const* d_in, const int* in_sizes, int n_in,
                              void* d_out, int out_size, void* d_ws, size_t ws_size,
                              hipStream_t stream)
{
    const float* x    = (const float*)d_in[0];   // [64][128][512]
    const float* Wi1  = (const float*)d_in[1];   // [512][2048]
    const float* A1   = (const float*)d_in[2];   // [2048][2048]
    const float* b1   = (const float*)d_in[3];   // [2048]
    const float* Wi2  = (const float*)d_in[4];
    const float* A2   = (const float*)d_in[5];
    const float* b2   = (const float*)d_in[6];
    const float* m01  = (const float*)d_in[7];   // [64][2048]
    const float* m02  = (const float*)d_in[8];
    float* out = (float*)d_out;
    char*  ws  = (char*)d_ws;

    unsigned short* xb  = (unsigned short*)(ws + XB_OFF);
    unsigned short* w1t = (unsigned short*)(ws + W1T_OFF);
    unsigned short* w2t = (unsigned short*)(ws + W2T_OFF);
    unsigned short* a1t = (unsigned short*)(ws + A1T_OFF);
    unsigned short* a2t = (unsigned short*)(ws + A2T_OFF);
    float*          Uf  = (float*)(ws + U_OFF);
    unsigned short* sb0 = (unsigned short*)(ws + S0_OFF);
    unsigned short* sb1 = (unsigned short*)(ws + S1_OFF);

    // 1) bf16 conversions / transposes (one-time per call)
    {
        int n = MROWS * N_IN;
        k_cvt<<<(n + 255) / 256, 256, 0, stream>>>(x, xb, n);
    }
    {
        int n = N_IN * N_HID;
        k_transpose<<<(n + 255) / 256, 256, 0, stream>>>(Wi1, w1t, N_IN, N_HID);
        k_transpose<<<(n + 255) / 256, 256, 0, stream>>>(Wi2, w2t, N_IN, N_HID);
    }
    {
        int n = N_HID * N_HID;
        k_transpose<<<(n + 255) / 256, 256, 0, stream>>>(A1, a1t, N_HID, N_HID);
        k_transpose<<<(n + 255) / 256, 256, 0, stream>>>(A2, a2t, N_HID, N_HID);
    }

    // 2) Feedforward drive U = x@W_in + b for both layers (WMMA GEMM)
    k_gemm_u<<<dim3(N_HID / 256, MROWS / 32, 2), 256, 0, stream>>>(
        xb, w1t, w2t, b1, b2, Uf);

    // 3) t=0 state
    {
        int n = 2 * BATCH * N_HID;
        k_init<<<(n + 255) / 256, 256, 0, stream>>>(m01, m02, out, sb0);
    }

    // 4) Recurrent scan: 128 stream-ordered WMMA steps, ping-pong spikes
    for (int t = 0; t < FRAMES; ++t) {
        unsigned short* si = (t & 1) ? sb1 : sb0;
        unsigned short* so = (t & 1) ? sb0 : sb1;
        k_step<<<dim3(N_HID / 64, 1, 2), 256, 0, stream>>>(
            a1t, a2t, Uf, si, so, out, t);
    }
}
